// EntropyCalculator_62972810494570
// MI455X (gfx1250) — matrix operand order repaired and checked by reading, not compiled, not run on hardware
//
#include <hip/hip_runtime.h>
#include <hip/hip_bf16.h>

typedef __attribute__((ext_vector_type(16))) _Float16 v16h;
typedef __attribute__((ext_vector_type(8)))  float    v8f;

#define DEVI static __device__ __forceinline__

// ---------------------------------------------------------------------------
// CDNA5 wave32 WMMA fragment loaders (16x16x32 f16, per CDNA5 ISA)
// A (16xK tile, MxK): lane<16: row=lane, K={k0..k0+7, k0+16..k0+23}
//                     lane>=16: row=lane-16, K={k0+8..k0+15, k0+24..k0+31}
// B (Kx16 tile as W[N,K] row-major): lane<16: col=lane,  K={k0..k0+7, k0+16..k0+23}
//                                    lane>=16: col=lane-16, K={k0+8..k0+15, k0+24..k0+31}
// C/D: VGPR r: lane<16 -> (row r,   col lane), lane>=16 -> (row r+8, col lane-16)
// ---------------------------------------------------------------------------
DEVI v16h load_frag_a(const _Float16* base, int stride, int row0, int k0, int lane) {
  int r  = row0 + (lane & 15);
  int kh = (lane >> 4) * 8;
  const _Float16* p = base + (size_t)r * stride + k0 + kh;
  v16h f;
  ((float4*)&f)[0] = *(const float4*)(p);
  ((float4*)&f)[1] = *(const float4*)(p + 16);
  return f;
}

DEVI v16h load_frag_b(const _Float16* w, int stride, int col0, int k0, int lane) {
  int n = col0 + (lane & 15);
  int k = k0 + ((lane >> 4) << 3);
  const _Float16* p = w + (size_t)n * stride + k;
  v16h f;
  ((float4*)&f)[0] = *(const float4*)(p);
  ((float4*)&f)[1] = *(const float4*)(p + 16);
  return f;
}

DEVI v8f wmma_f16(v16h a, v16h b, v8f c) {
  return __builtin_amdgcn_wmma_f32_16x16x32_f16(false, a, false, b, (short)0, c, false, false);
}

// ---------------------------------------------------------------------------
// f32 -> f16 weight conversion
// ---------------------------------------------------------------------------
__global__ void k_f32_to_f16(const float* __restrict__ src, _Float16* __restrict__ dst, int n) {
  int i = blockIdx.x * 256 + threadIdx.x;
  if (i < n) dst[i] = (_Float16)src[i];
}

// ---------------------------------------------------------------------------
// Embedding gather: x32[t,d] = emb[bytes[t],d]; also f16 copy for GEMM A
// ---------------------------------------------------------------------------
__global__ void k_embed(const int* __restrict__ bytes, const float* __restrict__ emb,
                        float* __restrict__ x32, _Float16* __restrict__ x16) {
  int t = blockIdx.x, d = threadIdx.x;             // grid 32768, block 128
  float v = emb[(size_t)bytes[t] * 128 + d];
  x32[(size_t)t * 128 + d] = v;
  x16[(size_t)t * 128 + d] = (_Float16)v;
}

// ---------------------------------------------------------------------------
// Generic GEMM: out[M,N] = A16[M,K] @ W16[N,K]^T + bias, optional relu.
// block=128 (4 waves); each wave computes a 32x32 tile via 2x2 register
// blocking (2 A-frags + 2 B-frags -> 4 WMMAs per K-step, 2x fragment reuse).
// grid = (M/128, N/32).
// ---------------------------------------------------------------------------
__global__ void __launch_bounds__(128) k_gemm_f16(
    const _Float16* __restrict__ A, const _Float16* __restrict__ W,
    const float* __restrict__ bias, float* __restrict__ out32,
    _Float16* __restrict__ out16, int N, int K, int relu)
{
  int lane = threadIdx.x & 31;
  int wave = threadIdx.x >> 5;
  int m0 = (blockIdx.x * 4 + wave) * 32;
  int n0 = blockIdx.y * 32;
  v8f c00 = {}, c01 = {}, c10 = {}, c11 = {};
  for (int k0 = 0; k0 < K; k0 += 32) {
    v16h a0 = load_frag_a(A, K, m0,      k0, lane);
    v16h a1 = load_frag_a(A, K, m0 + 16, k0, lane);
    v16h b0 = load_frag_b(W, K, n0,      k0, lane);
    v16h b1 = load_frag_b(W, K, n0 + 16, k0, lane);
    if (k0 + 32 < K)
      __builtin_prefetch(A + (size_t)(m0 + (lane & 15)) * K + k0 + 32, 0, 3);
    c00 = wmma_f16(a0, b0, c00);
    c01 = wmma_f16(a0, b1, c01);
    c10 = wmma_f16(a1, b0, c10);
    c11 = wmma_f16(a1, b1, c11);
  }
  int cl = lane & 15;
  int rb = (lane >> 4) << 3;
  float bv0 = bias[n0 + cl];
  float bv1 = bias[n0 + 16 + cl];
  #pragma unroll
  for (int r = 0; r < 8; ++r) {
    size_t row0 = (size_t)(m0 + rb + r);
    size_t row1 = (size_t)(m0 + 16 + rb + r);
    float v00 = c00[r] + bv0, v01 = c01[r] + bv1;
    float v10 = c10[r] + bv0, v11 = c11[r] + bv1;
    if (relu) {
      v00 = fmaxf(v00, 0.0f); v01 = fmaxf(v01, 0.0f);
      v10 = fmaxf(v10, 0.0f); v11 = fmaxf(v11, 0.0f);
    }
    if (out32) {
      out32[row0 * N + n0 + cl]      = v00;
      out32[row0 * N + n0 + 16 + cl] = v01;
      out32[row1 * N + n0 + cl]      = v10;
      out32[row1 * N + n0 + 16 + cl] = v11;
    }
    if (out16) {
      out16[row0 * N + n0 + cl]      = (_Float16)v00;
      out16[row0 * N + n0 + 16 + cl] = (_Float16)v01;
      out16[row1 * N + n0 + cl]      = (_Float16)v10;
      out16[row1 * N + n0 + 16 + cl] = (_Float16)v11;
    }
  }
}

// ---------------------------------------------------------------------------
// Repack qkv[M,384] -> q,k:[BH,S,32] and vT:[BH,32,S] so attention fragments
// are contiguous 128-bit loads.
// ---------------------------------------------------------------------------
__global__ void k_repack(const _Float16* __restrict__ qkv, _Float16* __restrict__ q,
                         _Float16* __restrict__ k, _Float16* __restrict__ vT) {
  int t = blockIdx.x, d = threadIdx.x;             // grid 32768, block 128
  int b = t >> 11, s = t & 2047;
  int h = d >> 5, dh = d & 31;
  int bh = b * 4 + h;
  const _Float16* row = qkv + (size_t)t * 384;
  q [((size_t)bh * 2048 + s) * 32 + dh] = row[d];
  k [((size_t)bh * 2048 + s) * 32 + dh] = row[128 + d];
  vT[((size_t)bh * 32 + dh) * 2048 + s] = row[256 + d];
}

// ---------------------------------------------------------------------------
// Flash attention: grid=(BH=64, S/64=32), block=128 (4 waves, each a 16-query
// tile). Online softmax, 32 keys per iteration. DH=32 -> single WMMA K-depth.
// Writes ctx directly as f16 into the [M,128] activation buffer.
// ---------------------------------------------------------------------------
__global__ void __launch_bounds__(128) k_attn(
    const _Float16* __restrict__ Q, const _Float16* __restrict__ Kd,
    const _Float16* __restrict__ Vt, _Float16* __restrict__ ctx_out)
{
  __shared__ _Float16 pshm[4][16][32];
  int lane = threadIdx.x & 31;
  int wave = threadIdx.x >> 5;
  int bh = blockIdx.x;
  int q0 = blockIdx.y * 64 + wave * 16;
  const _Float16* Qb = Q  + (size_t)bh * 2048 * 32;
  const _Float16* Kb = Kd + (size_t)bh * 2048 * 32;
  const _Float16* Vb = Vt + (size_t)bh * 32 * 2048;
  const float scale = 0.17677669529663689f;        // 1/sqrt(32)

  v16h qa = load_frag_a(Qb, 32, q0, 0, lane);
  v8f ctx0 = {}, ctx1 = {};
  v8f z = {};
  float Mr[8], Lr[8];
  #pragma unroll
  for (int r = 0; r < 8; ++r) { Mr[r] = -1e30f; Lr[r] = 0.0f; }

  int qrow = ((lane >> 4) << 3);                   // row base this lane-half touches

  for (int kt = 0; kt < 2048; kt += 32) {
    v16h kb0 = load_frag_b(Kb, 32, kt,      0, lane);
    v16h kb1 = load_frag_b(Kb, 32, kt + 16, 0, lane);
    v8f s0 = wmma_f16(qa, kb0, z);
    v8f s1 = wmma_f16(qa, kb1, z);

    #pragma unroll
    for (int r = 0; r < 8; ++r) {
      float a0 = s0[r] * scale, a1 = s1[r] * scale;
      float mx = fmaxf(a0, a1);                    // row reduce within 16-lane half
      mx = fmaxf(mx, __shfl_xor(mx, 1, 32));
      mx = fmaxf(mx, __shfl_xor(mx, 2, 32));
      mx = fmaxf(mx, __shfl_xor(mx, 4, 32));
      mx = fmaxf(mx, __shfl_xor(mx, 8, 32));
      float nm = fmaxf(Mr[r], mx);
      float alpha = __expf(Mr[r] - nm);
      float p0 = __expf(a0 - nm), p1 = __expf(a1 - nm);
      float ls = p0 + p1;
      ls += __shfl_xor(ls, 1, 32);
      ls += __shfl_xor(ls, 2, 32);
      ls += __shfl_xor(ls, 4, 32);
      ls += __shfl_xor(ls, 8, 32);
      Lr[r] = Lr[r] * alpha + ls;
      Mr[r] = nm;
      ctx0[r] *= alpha;
      ctx1[r] *= alpha;
      pshm[wave][qrow + r][(lane & 15)]      = (_Float16)p0;   // re-layout P via LDS
      pshm[wave][qrow + r][(lane & 15) + 16] = (_Float16)p1;
    }
    __syncthreads();
    v16h pa  = load_frag_a(&pshm[wave][0][0], 32, 0, 0, lane);
    v16h vb0 = load_frag_b(Vb, 2048, 0,  kt, lane);
    v16h vb1 = load_frag_b(Vb, 2048, 16, kt, lane);
    ctx0 = wmma_f16(pa, vb0, ctx0);
    ctx1 = wmma_f16(pa, vb1, ctx1);
    __syncthreads();
  }

  int b = bh >> 2, h = bh & 3;
  #pragma unroll
  for (int r = 0; r < 8; ++r) {
    size_t t = (size_t)b * 2048 + (q0 + qrow + r);
    float inv = 1.0f / Lr[r];
    ctx_out[t * 128 + h * 32 + (lane & 15)]      = (_Float16)(ctx0[r] * inv);
    ctx_out[t * 128 + h * 32 + (lane & 15) + 16] = (_Float16)(ctx1[r] * inv);
  }
}

// ---------------------------------------------------------------------------
// x = LN(x + y) * g + b ; writes f32 (residual stream) and f16 (next GEMM A)
// ---------------------------------------------------------------------------
__global__ void k_res_ln(float* __restrict__ X32, const float* __restrict__ y,
                         const float* __restrict__ g, const float* __restrict__ bta,
                         _Float16* __restrict__ x16) {
  __shared__ float red[128];
  int t = blockIdx.x, d = threadIdx.x;             // grid 32768, block 128
  float v = X32[(size_t)t * 128 + d] + y[(size_t)t * 128 + d];
  red[d] = v; __syncthreads();
  for (int o = 64; o > 0; o >>= 1) { if (d < o) red[d] += red[d + o]; __syncthreads(); }
  float mu = red[0] * (1.0f / 128.0f); __syncthreads();
  float dv = v - mu;
  red[d] = dv * dv; __syncthreads();
  for (int o = 64; o > 0; o >>= 1) { if (d < o) red[d] += red[d + o]; __syncthreads(); }
  float var = red[0] * (1.0f / 128.0f);
  float out = dv * rsqrtf(var + 1e-5f) * g[d] + bta[d];
  X32[(size_t)t * 128 + d] = out;
  x16[(size_t)t * 128 + d] = (_Float16)out;
}

// ---------------------------------------------------------------------------
// Per-token softmax entropy over 256 logits
// ---------------------------------------------------------------------------
__global__ void k_entropy(const float* __restrict__ logits, float* __restrict__ ent) {
  __shared__ float red[256];
  int t = blockIdx.x, i = threadIdx.x;             // grid 32768, block 256
  float v = logits[(size_t)t * 256 + i];
  red[i] = v; __syncthreads();
  for (int o = 128; o > 0; o >>= 1) { if (i < o) red[i] = fmaxf(red[i], red[i + o]); __syncthreads(); }
  float mx = red[0]; __syncthreads();
  float e = __expf(v - mx);
  red[i] = e; __syncthreads();
  for (int o = 128; o > 0; o >>= 1) { if (i < o) red[i] += red[i + o]; __syncthreads(); }
  float Z = red[0]; __syncthreads();
  float p = e / Z;
  red[i] = -p * __logf(p + 1e-10f); __syncthreads();
  for (int o = 128; o > 0; o >>= 1) { if (i < o) red[i] += red[i + o]; __syncthreads(); }
  if (i == 0) ent[t] = red[0];
}

__global__ void k_avg(const float* __restrict__ ent, float* __restrict__ avg) {
  int s = blockIdx.x * 256 + threadIdx.x;          // grid 8, block 256
  if (s < 2048) {
    float a = 0.0f;
    for (int b = 0; b < 16; ++b) a += ent[(size_t)b * 2048 + s];
    avg[s] = a * (1.0f / 16.0f);
  }
}

__global__ void k_patch(const float* __restrict__ avg, float* __restrict__ pid) {
  if (threadIdx.x == 0 && blockIdx.x == 0) {
    int acc = 0;
    for (int i = 0; i < 2048; ++i) {
      if (i > 0 && avg[i] > 4.0f) acc++;
      pid[i] = (float)acc;
    }
  }
}

// ---------------------------------------------------------------------------
extern "C" void kernel_launch(void* const* d_in, const int* in_sizes, int n_in,
                              void* d_out, int out_size, void* d_ws, size_t ws_size,
                              hipStream_t stream)
{
  (void)in_sizes; (void)n_in; (void)out_size; (void)ws_size;
  const int M = 32768;                             // B*S tokens
  const int* bytes = (const int*)d_in[0];
  const float* emb = (const float*)d_in[1];
  auto f = [&](int i) { return (const float*)d_in[i]; };

  // workspace arena (everything fits comfortably in the 192MB L2)
  char* ws = (char*)d_ws;
  size_t off = 0;
  auto alloc = [&](size_t nbytes) {
    off = (off + 255) & ~(size_t)255;
    void* p = ws + off;
    off += nbytes;
    return p;
  };
  float*    X32   = (float*)   alloc((size_t)M * 128 * 4);   // residual stream
  _Float16* A16   = (_Float16*)alloc((size_t)M * 128 * 2);   // f16 activations / ctx
  _Float16* QKV16 = (_Float16*)alloc((size_t)M * 384 * 2);
  _Float16* Q16   = (_Float16*)alloc((size_t)M * 128 * 2);
  _Float16* K16   = (_Float16*)alloc((size_t)M * 128 * 2);
  _Float16* VT16  = (_Float16*)alloc((size_t)M * 128 * 2);
  _Float16* FF16  = (_Float16*)alloc((size_t)M * 512 * 2);
  float*    C32   = (float*)   alloc((size_t)M * 256 * 4);   // f32 GEMM outputs
  float*    ENT   = (float*)   alloc((size_t)M * 4);
  _Float16* wqkv16[2], *wo16[2], *w1_16[2], *w2_16[2];
  for (int l = 0; l < 2; ++l) {
    wqkv16[l] = (_Float16*)alloc(384 * 128 * 2);
    wo16[l]   = (_Float16*)alloc(128 * 128 * 2);
    w1_16[l]  = (_Float16*)alloc(512 * 128 * 2);
    w2_16[l]  = (_Float16*)alloc(128 * 512 * 2);
  }
  _Float16* outw16 = (_Float16*)alloc(256 * 128 * 2);

  auto cvt = [&](const float* src, _Float16* dst, int n) {
    k_f32_to_f16<<<(n + 255) / 256, 256, 0, stream>>>(src, dst, n);
  };
  for (int l = 0; l < 2; ++l) {
    int base = 2 + l * 12;
    cvt(f(base + 0), wqkv16[l], 384 * 128);
    cvt(f(base + 2), wo16[l],   128 * 128);
    cvt(f(base + 6), w1_16[l],  512 * 128);
    cvt(f(base + 8), w2_16[l],  128 * 512);
  }
  cvt(f(26), outw16, 256 * 128);

  k_embed<<<M, 128, 0, stream>>>(bytes, emb, X32, A16);

  for (int l = 0; l < 2; ++l) {
    int base = 2 + l * 12;
    const float* bqkv = f(base + 1);
    const float* bo   = f(base + 3);
    const float* ln1g = f(base + 4);
    const float* ln1b = f(base + 5);
    const float* b1   = f(base + 7);
    const float* b2   = f(base + 9);
    const float* ln2g = f(base + 10);
    const float* ln2b = f(base + 11);

    k_gemm_f16<<<dim3(256, 12), 128, 0, stream>>>(A16, wqkv16[l], bqkv, nullptr, QKV16, 384, 128, 0);
    k_repack<<<M, 128, 0, stream>>>(QKV16, Q16, K16, VT16);
    k_attn<<<dim3(64, 32), 128, 0, stream>>>(Q16, K16, VT16, A16);
    k_gemm_f16<<<dim3(256, 4), 128, 0, stream>>>(A16, wo16[l], bo, C32, nullptr, 128, 128, 0);
    k_res_ln<<<M, 128, 0, stream>>>(X32, C32, ln1g, ln1b, A16);
    k_gemm_f16<<<dim3(256, 16), 128, 0, stream>>>(A16, w1_16[l], b1, nullptr, FF16, 512, 128, 1);
    k_gemm_f16<<<dim3(256, 4), 128, 0, stream>>>(FF16, w2_16[l], b2, C32, nullptr, 128, 512, 0);
    k_res_ln<<<M, 128, 0, stream>>>(X32, C32, ln2g, ln2b, A16);
  }

  k_gemm_f16<<<dim3(256, 8), 128, 0, stream>>>(A16, outw16, f(27), C32, nullptr, 256, 128, 0);
  k_entropy<<<M, 256, 0, stream>>>(C32, ENT);
  float* out = (float*)d_out;
  k_avg<<<dim3(8), dim3(256), 0, stream>>>(ENT, out);
  k_patch<<<1, 1, 0, stream>>>(out, out + 2048);
}
